// JSONTreeLSTM_26199300505763
// MI455X (gfx1250) — compile-verified, matrix operand order checked by reading, not run on hardware
//
#include <hip/hip_runtime.h>
#include <hip/hip_bf16.h>
#include <math.h>

// JSON tree LSTM, MI455X (gfx1250, wave32, WMMA bf16 16x16x32, f32 acc).
// K=64, L=128, F=16, S=32, M=128.

typedef __bf16 v8bf  __attribute__((ext_vector_type(8)));
typedef __bf16 v16bf __attribute__((ext_vector_type(16)));
typedef float  v8f   __attribute__((ext_vector_type(8)));

__device__ __forceinline__ float sigm(float x) { return 1.0f / (1.0f + __expf(-x)); }

// Load a 16x32 bf16 A (or 32x16 B) fragment for v_wmma_f32_16x16x32_bf16.
// Per ISA layout: lane&15 = row (A: M, B: N); lane>>4 selects K halves:
//   khalf=0 -> K = kbase+[0..7] and kbase+16+[0..7]
//   khalf=1 -> K = kbase+8+[0..7] and kbase+24+[0..7]
__device__ __forceinline__ v16bf load_frag(const __bf16* base, int row, int ld,
                                           int kbase, int khalf) {
  const __bf16* p = base + row * ld + kbase + khalf * 8;
  v8bf lo = *(const v8bf*)(p);
  v8bf hi = *(const v8bf*)(p + 16);
  v16bf r;
#pragma unroll
  for (int i = 0; i < 8; ++i) { r[i] = lo[i]; r[8 + i] = hi[i]; }
  return r;
}

__device__ __forceinline__ v8f wmma_bf16(v16bf a, v16bf b, v8f c) {
  return __builtin_amdgcn_wmma_f32_16x16x32_bf16(false, a, false, b, (short)0, c,
                                                 false, false);
}

// ---------------------------------------------------------------- prep: f32->bf16
__global__ void cvt_bf16(const float* __restrict__ src, __bf16* __restrict__ dst, int n) {
  int i = blockIdx.x * blockDim.x + threadIdx.x;
  if (i < n) dst[i] = (__bf16)src[i];
}

// ------------------------------------------------- string-leaf LSTM (S=32 steps)
// One block = 16 batch rows of KL=8192; 8 waves; wave w owns gate cols [64w,64w+64).
__global__ void __launch_bounds__(256, 1)
str_lstm(const int* __restrict__ chars, const float* __restrict__ embed,
         const __bf16* __restrict__ Wih, const __bf16* __restrict__ Whh,
         const float* __restrict__ bih, const float* __restrict__ bhh,
         float* __restrict__ h_out) {
  __shared__ __align__(16) __bf16 x_s[16][128];
  __shared__ __align__(16) __bf16 h_s[16][128];
  __shared__ float c_s[16][128];
  __shared__ float g_s[16][512];
  const int tid = threadIdx.x;
  const int lane = tid & 31, wave = tid >> 5;
  const int mrow = lane & 15, khalf = lane >> 4;
  const int row0 = blockIdx.x * 16;

  for (int e = tid; e < 16 * 128; e += 256) {
    int r = e >> 7, m = e & 127;
    h_s[r][m] = (__bf16)0.0f;
    c_s[r][m] = 0.0f;
  }
  // warm L2 path for this wave's weight rows
  __builtin_prefetch(Wih + (wave * 64 + mrow) * 128, 0, 1);
  __builtin_prefetch(Whh + (wave * 64 + mrow) * 128, 0, 1);
  __syncthreads();

  for (int t = 0; t < 32; ++t) {
    // embedding lookup for this time step
    for (int e = tid; e < 16 * 128; e += 256) {
      int r = e >> 7, m = e & 127;
      int ch = chars[(row0 + r) * 32 + t];
      x_s[r][m] = (__bf16)embed[ch * 128 + m];
    }
    __syncthreads();
#pragma unroll
    for (int j = 0; j < 4; ++j) {
      int nt = wave * 4 + j;
      v8f acc = {};
#pragma unroll
      for (int kt = 0; kt < 4; ++kt) {
        v16bf a = load_frag(&x_s[0][0], mrow, 128, kt * 32, khalf);
        v16bf b = load_frag(Wih, nt * 16 + mrow, 128, kt * 32, khalf);
        acc = wmma_bf16(a, b, acc);
      }
#pragma unroll
      for (int kt = 0; kt < 4; ++kt) {
        v16bf a = load_frag(&h_s[0][0], mrow, 128, kt * 32, khalf);
        v16bf b = load_frag(Whh, nt * 16 + mrow, 128, kt * 32, khalf);
        acc = wmma_bf16(a, b, acc);
      }
#pragma unroll
      for (int v = 0; v < 8; ++v)
        g_s[v + 8 * khalf][nt * 16 + mrow] = acc[v];
    }
    __syncthreads();
    // gate order i,f,g,o
    for (int e = tid; e < 16 * 128; e += 256) {
      int r = e >> 7, m = e & 127;
      float gi = g_s[r][m]         + bih[m]         + bhh[m];
      float gf = g_s[r][128 + m]   + bih[128 + m]   + bhh[128 + m];
      float gg = g_s[r][256 + m]   + bih[256 + m]   + bhh[256 + m];
      float go = g_s[r][384 + m]   + bih[384 + m]   + bhh[384 + m];
      float c = sigm(gf) * c_s[r][m] + sigm(gi) * tanhf(gg);
      c_s[r][m] = c;
      h_s[r][m] = (__bf16)(sigm(go) * tanhf(c));
    }
    __syncthreads();
  }
  for (int e = tid; e < 16 * 128; e += 256) {
    int r = e >> 7, m = e & 127;
    h_out[(row0 + r) * 128 + m] = (float)h_s[r][m];
  }
}

// ------------------------------------------------------ element-object node
// hs_bar = sumF(numbers)*W_num + 16*b_num + h_str ; leaf c==0 so fc term is 0.
// iou GEMM (128->384), pointwise, then hhat = h_e @ Wlout.T + blout.
__global__ void __launch_bounds__(256, 1)
elem_node(const float* __restrict__ numbers, const float* __restrict__ Wnum,
          const float* __restrict__ bnum, const float* __restrict__ h_str,
          const __bf16* __restrict__ Wiou, const float* __restrict__ biou,
          const __bf16* __restrict__ Wlout, const float* __restrict__ blout,
          float* __restrict__ c_e, float* __restrict__ hhat_e) {
  __shared__ __align__(16) __bf16 hs_s[16][128];
  __shared__ __align__(16) __bf16 he_s[16][128];
  __shared__ float g_s[16][384];
  __shared__ float sumF[16];
  const int tid = threadIdx.x;
  const int lane = tid & 31, wave = tid >> 5;
  const int mrow = lane & 15, khalf = lane >> 4;
  const int row0 = blockIdx.x * 16;

  if (tid < 16) {
    float s = 0.0f;
    for (int f = 0; f < 16; ++f) s += numbers[(row0 + tid) * 16 + f];
    sumF[tid] = s;
  }
  __syncthreads();
  for (int e = tid; e < 16 * 128; e += 256) {
    int r = e >> 7, m = e & 127;
    hs_s[r][m] = (__bf16)(sumF[r] * Wnum[m] + 16.0f * bnum[m] +
                          h_str[(row0 + r) * 128 + m]);
  }
  __syncthreads();
  // iou GEMM: 384 cols = 24 N-tiles, 3 per wave
#pragma unroll
  for (int j = 0; j < 3; ++j) {
    int nt = wave * 3 + j;
    v8f acc = {};
#pragma unroll
    for (int kt = 0; kt < 4; ++kt) {
      v16bf a = load_frag(&hs_s[0][0], mrow, 128, kt * 32, khalf);
      v16bf b = load_frag(Wiou, nt * 16 + mrow, 128, kt * 32, khalf);
      acc = wmma_bf16(a, b, acc);
    }
#pragma unroll
    for (int v = 0; v < 8; ++v)
      g_s[v + 8 * khalf][nt * 16 + mrow] = acc[v];
  }
  __syncthreads();
  // gate order i,o,u
  for (int e = tid; e < 16 * 128; e += 256) {
    int r = e >> 7, m = e & 127;
    float gi = g_s[r][m] + biou[m];
    float go = g_s[r][128 + m] + biou[128 + m];
    float gu = g_s[r][256 + m] + biou[256 + m];
    float c = sigm(gi) * tanhf(gu);
    c_e[(row0 + r) * 128 + m] = c;
    he_s[r][m] = (__bf16)(sigm(go) * tanhf(c));
  }
  __syncthreads();
  // hhat GEMM: 128 cols = 8 N-tiles, 1 per wave
  {
    int nt = wave;
    v8f acc = {};
#pragma unroll
    for (int kt = 0; kt < 4; ++kt) {
      v16bf a = load_frag(&he_s[0][0], mrow, 128, kt * 32, khalf);
      v16bf b = load_frag(Wlout, nt * 16 + mrow, 128, kt * 32, khalf);
      acc = wmma_bf16(a, b, acc);
    }
#pragma unroll
    for (int v = 0; v < 8; ++v) {
      int rr = v + 8 * khalf;
      int n = nt * 16 + mrow;
      hhat_e[(row0 + rr) * 128 + n] = acc[v] + blout[n];
    }
  }
}

// ----------------------------------------------------- array LSTM (L=128 steps)
// batch = K = 64 arrays; input = cat([c_e, hhat_e]) (256 wide).
__global__ void __launch_bounds__(256, 1)
arr_lstm(const float* __restrict__ c_e, const float* __restrict__ hhat_e,
         const __bf16* __restrict__ Wih, const __bf16* __restrict__ Whh,
         const float* __restrict__ bih, const float* __restrict__ bhh,
         float* __restrict__ arr_c, float* __restrict__ arr_h) {
  __shared__ __align__(16) __bf16 x_s[16][256];
  __shared__ __align__(16) __bf16 h_s[16][128];
  __shared__ float c_s[16][128];
  __shared__ float g_s[16][512];
  const int tid = threadIdx.x;
  const int lane = tid & 31, wave = tid >> 5;
  const int mrow = lane & 15, khalf = lane >> 4;
  const int row0 = blockIdx.x * 16;  // k-index base

  for (int e = tid; e < 16 * 128; e += 256) {
    int r = e >> 7, m = e & 127;
    h_s[r][m] = (__bf16)0.0f;
    c_s[r][m] = 0.0f;
  }
  __syncthreads();

  for (int t = 0; t < 128; ++t) {
    for (int e = tid; e < 16 * 256; e += 256) {
      int r = e >> 8, col = e & 255;
      int k = row0 + r;
      float v = (col < 128) ? c_e[(k * 128 + t) * 128 + col]
                            : hhat_e[(k * 128 + t) * 128 + (col - 128)];
      x_s[r][col] = (__bf16)v;
    }
    __syncthreads();
#pragma unroll
    for (int j = 0; j < 4; ++j) {
      int nt = wave * 4 + j;
      v8f acc = {};
#pragma unroll
      for (int kt = 0; kt < 8; ++kt) {
        v16bf a = load_frag(&x_s[0][0], mrow, 256, kt * 32, khalf);
        v16bf b = load_frag(Wih, nt * 16 + mrow, 256, kt * 32, khalf);
        acc = wmma_bf16(a, b, acc);
      }
#pragma unroll
      for (int kt = 0; kt < 4; ++kt) {
        v16bf a = load_frag(&h_s[0][0], mrow, 128, kt * 32, khalf);
        v16bf b = load_frag(Whh, nt * 16 + mrow, 128, kt * 32, khalf);
        acc = wmma_bf16(a, b, acc);
      }
#pragma unroll
      for (int v = 0; v < 8; ++v)
        g_s[v + 8 * khalf][nt * 16 + mrow] = acc[v];
    }
    __syncthreads();
    for (int e = tid; e < 16 * 128; e += 256) {
      int r = e >> 7, m = e & 127;
      float gi = g_s[r][m]         + bih[m]         + bhh[m];
      float gf = g_s[r][128 + m]   + bih[128 + m]   + bhh[128 + m];
      float gg = g_s[r][256 + m]   + bih[256 + m]   + bhh[256 + m];
      float go = g_s[r][384 + m]   + bih[384 + m]   + bhh[384 + m];
      float c = sigm(gf) * c_s[r][m] + sigm(gi) * tanhf(gg);
      c_s[r][m] = c;
      h_s[r][m] = (__bf16)(sigm(go) * tanhf(c));
    }
    __syncthreads();
  }
  // reference swaps: arr_c = hidden, arr_h = cell
  for (int e = tid; e < 16 * 128; e += 256) {
    int r = e >> 7, m = e & 127;
    int k = row0 + r;
    arr_c[k * 128 + m] = (float)h_s[r][m];
    arr_h[k * 128 + m] = c_s[r][m];
  }
}

// ------------------------------------------------------------- root object node
__global__ void __launch_bounds__(256, 1)
root_node(const float* __restrict__ arr_c, const float* __restrict__ arr_h,
          const float* __restrict__ Wfh, const float* __restrict__ bfh,
          const float* __restrict__ Wiou, const float* __restrict__ biou,
          const float* __restrict__ Wlout, const float* __restrict__ blout,
          float* __restrict__ out) {
  __shared__ float fcp[2][128], hsp[2][128];
  __shared__ float fc[128], hs[128];
  __shared__ float g[384];
  __shared__ float hr[128];
  const int tid = threadIdx.x;
  const int m = tid & 127, grp = tid >> 7;

  float facc = 0.0f, hacc = 0.0f;
  for (int k = grp * 32; k < grp * 32 + 32; ++k) {
    float dot = bfh[m];
    for (int j = 0; j < 128; ++j) dot += Wfh[m * 128 + j] * arr_h[k * 128 + j];
    facc += sigm(dot) * arr_c[k * 128 + m];
    hacc += arr_h[k * 128 + m];
  }
  fcp[grp][m] = facc;
  hsp[grp][m] = hacc;
  __syncthreads();
  if (tid < 128) {
    fc[m] = fcp[0][m] + fcp[1][m];
    hs[m] = hsp[0][m] + hsp[1][m];
  }
  __syncthreads();
  for (int idx = tid; idx < 384; idx += 256) {
    float dot = biou[idx];
    for (int j = 0; j < 128; ++j) dot += Wiou[idx * 128 + j] * hs[j];
    g[idx] = dot;
  }
  __syncthreads();
  if (tid < 128) {
    float c = sigm(g[m]) * tanhf(g[256 + m]) + fc[m];  // i,o,u split
    hr[m] = sigm(g[128 + m]) * tanhf(c);
    out[m] = c;
  }
  __syncthreads();
  if (tid < 128) {
    float dot = blout[m];
    for (int j = 0; j < 128; ++j) dot += Wlout[m * 128 + j] * hr[j];
    out[128 + m] = dot;
  }
}

extern "C" void kernel_launch(void* const* d_in, const int* in_sizes, int n_in,
                              void* d_out, int out_size, void* d_ws, size_t ws_size,
                              hipStream_t stream) {
  (void)in_sizes; (void)n_in; (void)out_size; (void)ws_size;
  const float* numbers = (const float*)d_in[0];
  const int*   chars   = (const int*)d_in[1];
  const float* Wnum    = (const float*)d_in[2];
  const float* bnum    = (const float*)d_in[3];
  const float* embed   = (const float*)d_in[4];
  const float* Wih_s   = (const float*)d_in[5];
  const float* Whh_s   = (const float*)d_in[6];
  const float* bih_s   = (const float*)d_in[7];
  const float* bhh_s   = (const float*)d_in[8];
  const float* Wih_a   = (const float*)d_in[9];
  const float* Whh_a   = (const float*)d_in[10];
  const float* bih_a   = (const float*)d_in[11];
  const float* bhh_a   = (const float*)d_in[12];
  const float* Wfh     = (const float*)d_in[13];
  const float* bfh     = (const float*)d_in[14];
  const float* Wiou    = (const float*)d_in[15];
  const float* biou    = (const float*)d_in[16];
  const float* Wlout   = (const float*)d_in[17];
  const float* blout   = (const float*)d_in[18];

  char* ws = (char*)d_ws;
  __bf16* WihS_b  = (__bf16*)(ws + 0);         // 512*128*2
  __bf16* WhhS_b  = (__bf16*)(ws + 131072);    // 512*128*2
  __bf16* WihA_b  = (__bf16*)(ws + 262144);    // 512*256*2
  __bf16* WhhA_b  = (__bf16*)(ws + 524288);    // 512*128*2
  __bf16* Wiou_b  = (__bf16*)(ws + 655360);    // 384*128*2
  __bf16* Wlout_b = (__bf16*)(ws + 753664);    // 128*128*2
  float* h_str  = (float*)(ws + 786432);       // 8192*128*4
  float* c_e    = (float*)(ws + 4980736);      // 8192*128*4
  float* hhat_e = (float*)(ws + 9175040);      // 8192*128*4
  float* arr_c  = (float*)(ws + 13369344);     // 64*128*4
  float* arr_h  = (float*)(ws + 13402112);     // 64*128*4

  cvt_bf16<<<(65536 + 255) / 256, 256, 0, stream>>>(Wih_s, WihS_b, 65536);
  cvt_bf16<<<(65536 + 255) / 256, 256, 0, stream>>>(Whh_s, WhhS_b, 65536);
  cvt_bf16<<<(131072 + 255) / 256, 256, 0, stream>>>(Wih_a, WihA_b, 131072);
  cvt_bf16<<<(65536 + 255) / 256, 256, 0, stream>>>(Whh_a, WhhA_b, 65536);
  cvt_bf16<<<(49152 + 255) / 256, 256, 0, stream>>>(Wiou, Wiou_b, 49152);
  cvt_bf16<<<(16384 + 255) / 256, 256, 0, stream>>>(Wlout, Wlout_b, 16384);

  str_lstm<<<512, 256, 0, stream>>>(chars, embed, WihS_b, WhhS_b, bih_s, bhh_s, h_str);
  elem_node<<<512, 256, 0, stream>>>(numbers, Wnum, bnum, h_str, Wiou_b, biou,
                                     Wlout_b, blout, c_e, hhat_e);
  arr_lstm<<<4, 256, 0, stream>>>(c_e, hhat_e, WihA_b, WhhA_b, bih_a, bhh_a,
                                  arr_c, arr_h);
  root_node<<<1, 256, 0, stream>>>(arr_c, arr_h, Wfh, bfh, Wiou, biou, Wlout,
                                   blout, (float*)d_out);
}
